// MultiHeadLatentAttention_1537598292081
// MI455X (gfx1250) — matrix operand order repaired and checked by reading, not compiled, not run on hardware
//
#include <hip/hip_runtime.h>

#define B_SZ   2
#define SEQ    2048
#define DMODEL 2048
#define NHEADS 16
#define DLAT   512
#define DK     128
#define MROWS  (B_SZ * SEQ) /* 4096 */

typedef __attribute__((ext_vector_type(8)))  __bf16 v8bf;
typedef __attribute__((ext_vector_type(16))) __bf16 v16bf;
typedef __attribute__((ext_vector_type(8)))  float  v8f;
typedef __attribute__((ext_vector_type(4)))  unsigned int v4u;
typedef __attribute__((ext_vector_type(8)))  int    v8i;
typedef __attribute__((ext_vector_type(4)))  int    v4i;

// ---- WMMA fragment helpers --------------------------------------------------
// A-fragment (16x32, MxK): lane l row m=l&15; elem i -> k=16*(l>>4)+i (contiguous).
// B-fragment (32x16, KxN): lane l col n=l&15; elem i -> k=16*(l>>4)+i (from row n of
// a pre-transposed N x K matrix => contiguous).
static __device__ __forceinline__ v16bf cat16(v8bf lo, v8bf hi) {
  return __builtin_shufflevector(lo, hi, 0,1,2,3,4,5,6,7,8,9,10,11,12,13,14,15);
}
static __device__ __forceinline__ v16bf ldfrag_a(const __bf16* p) {
  v8bf lo = *(const v8bf*)p;
  v8bf hi = *(const v8bf*)(p + 8);
  return cat16(lo, hi);
}
static __device__ __forceinline__ v16bf ldfrag_b(const __bf16* p) {
  v8bf lo = *(const v8bf*)p;
  v8bf hi = *(const v8bf*)(p + 8);
  return cat16(lo, hi);
}

// ---- Tensor Data Mover: 2-D tile load, global -> LDS ------------------------
// D# per CDNA5 ISA. bf16 elements (data_size code 1).
// pad_enable inserts (pad_amt_code+1) DWORDs of LDS padding after each
// 2^(pad_int_code+1) DWORDs stored => bank-conflict-free padded rows.
static __device__ __forceinline__ unsigned int lds_off_u32(const void* p) {
  // LDS aperture: addr[31:0] is the LDS byte offset (ISA §10.2).
  return (unsigned int)(unsigned long long)(uintptr_t)p;
}
static __device__ __forceinline__ void tdm_load_2d(
    unsigned int lds_off, const void* gsrc, unsigned int dim0, unsigned int rows,
    unsigned int stride_elems, unsigned int pad_int_code, unsigned int pad_amt_code) {
  unsigned long long ga = (unsigned long long)(uintptr_t)gsrc;
  v4u g0;
  g0.x = 1u;                                   // count=1, user mode, no gather
  g0.y = lds_off;                              // lds_addr
  g0.z = (unsigned int)ga;                     // global_addr[31:0]
  g0.w = (unsigned int)((ga >> 32) & 0x01ffffffu) | 0x80000000u; // addr[56:32]|type=2
  v8i g1;
  g1[0] = (int)((1u << 16) |                   // data_size = 2 bytes
                (1u << 20) |                   // pad_enable
                (pad_int_code << 22) | (pad_amt_code << 25));
  g1[1] = (int)((dim0 & 0xffffu) << 16);                       // tensor_dim0 lo16
  g1[2] = (int)(((dim0 >> 16) & 0xffffu) | ((rows & 0xffffu) << 16)); // dim0 hi|dim1 lo
  g1[3] = (int)(((rows >> 16) & 0xffffu) | ((dim0 & 0xffffu) << 16)); // dim1 hi|tile_dim0
  g1[4] = (int)(rows & 0xffffu);                               // tile_dim1 | tile_dim2=0
  g1[5] = (int)stride_elems;                                   // tensor_dim0_stride lo32
  g1[6] = 0;
  g1[7] = 0;
  v4i z4 = {0, 0, 0, 0};
  v8i z8 = {0, 0, 0, 0, 0, 0, 0, 0};
  // amdgpu-toolchain (clang-23) 6-arg form: (g0, g1, g2, g3, g4, cpol)
  __builtin_amdgcn_tensor_load_to_lds(g0, g1, z4, z4, z8, 0);
}

// ---- conversion kernels -----------------------------------------------------
__global__ void cvt_bf16_kernel(const float* __restrict__ src, __bf16* __restrict__ dst, int n) {
  int i = blockIdx.x * 256 + threadIdx.x;
  if (i < n) dst[i] = (__bf16)src[i];
}

// src K x N fp32 -> dst N x K bf16 (= src^T), 32x32 LDS-tiled, coalesced both sides
__global__ __launch_bounds__(256)
void cvt_t_bf16_kernel(const float* __restrict__ src, __bf16* __restrict__ dst, int K, int N) {
  __shared__ __bf16 t[32][33];
  const int n0 = blockIdx.x * 32, k0 = blockIdx.y * 32;
  const int tx = threadIdx.x & 31, ty = threadIdx.x >> 5;   // ty: 0..7
#pragma unroll
  for (int r = 0; r < 4; ++r)
    t[ty + r * 8][tx] = (__bf16)src[(size_t)(k0 + ty + r * 8) * N + n0 + tx];
  __syncthreads();
#pragma unroll
  for (int r = 0; r < 4; ++r)
    dst[(size_t)(n0 + ty + r * 8) * K + k0 + tx] = t[tx][ty + r * 8];
}

// V [B*S][DMODEL] bf16 -> Vt [B][H][DK][S] bf16, 32x32 LDS-tiled
__global__ __launch_bounds__(256)
void transpose_v_kernel(const __bf16* __restrict__ V, __bf16* __restrict__ Vt) {
  __shared__ __bf16 t[32][33];
  const int b = blockIdx.z;
  const int h = blockIdx.y >> 2, dt = blockIdx.y & 3;
  const int s0 = blockIdx.x * 32, d0 = dt * 32;
  const int tx = threadIdx.x & 31, ty = threadIdx.x >> 5;
#pragma unroll
  for (int r = 0; r < 4; ++r)
    t[ty + r * 8][tx] = V[(size_t)(b * SEQ + s0 + ty + r * 8) * DMODEL + h * DK + d0 + tx];
  __syncthreads();
#pragma unroll
  for (int r = 0; r < 4; ++r)
    Vt[(size_t)((b * NHEADS + h) * DK + d0 + ty + r * 8) * SEQ + s0 + tx] = t[tx][ty + r * 8];
}

// ---- LDS-tiled bf16 WMMA GEMM: C(MxN) = A(MxK) @ Bt(NxK)^T + bias ----------
// Block = 256 thr (8 waves) computes 128x128; TDM-staged, double-buffered LDS.
#define LSTR 40   // padded LDS row stride (32 bf16 data + 8 pad = 80B) -> conflict-free
template <typename OutT>
__global__ __launch_bounds__(256)
void gemm_bf16_wmma(const __bf16* __restrict__ A, const __bf16* __restrict__ Bt,
                    const float* __restrict__ bias, OutT* __restrict__ C,
                    int M, int N, int K) {
  __shared__ __align__(16) __bf16 Abuf[2][128 * LSTR];
  __shared__ __align__(16) __bf16 Bbuf[2][128 * LSTR];
  const int lane = threadIdx.x & 31;
  const int wave = threadIdx.x >> 5;
  const int l15 = lane & 15, lh = lane >> 4;
  const int wm = wave >> 1, wn = wave & 1;     // wave tile: 32 rows x 64 cols
  const int tm = blockIdx.y, tn = blockIdx.x;

  const __bf16* Ablk = A  + (size_t)(tm * 128) * K;
  const __bf16* Bblk = Bt + (size_t)(tn * 128) * K;

  v8f acc[2][4];
  v8f vz = {};
#pragma unroll
  for (int mt = 0; mt < 2; ++mt)
#pragma unroll
    for (int nt = 0; nt < 4; ++nt) acc[mt][nt] = vz;

  if (wave == 0) {   // pad: 16 DW interval (code 3), 4 DW amount (code 3)
    tdm_load_2d(lds_off_u32(&Abuf[0][0]), Ablk, 32, 128, (unsigned)K, 3, 3);
    tdm_load_2d(lds_off_u32(&Bbuf[0][0]), Bblk, 32, 128, (unsigned)K, 3, 3);
  }
  int buf = 0;
  for (int k0 = 0; k0 < K; k0 += 32) {
    if (wave == 0) __builtin_amdgcn_s_wait_tensorcnt(0);
    __syncthreads();
    if (wave == 0 && k0 + 32 < K) {
      tdm_load_2d(lds_off_u32(&Abuf[buf ^ 1][0]), Ablk + (k0 + 32), 32, 128, (unsigned)K, 3, 3);
      tdm_load_2d(lds_off_u32(&Bbuf[buf ^ 1][0]), Bblk + (k0 + 32), 32, 128, (unsigned)K, 3, 3);
    }
    v16bf af[2], bfr[4];
#pragma unroll
    for (int mt = 0; mt < 2; ++mt)
      af[mt] = ldfrag_a(&Abuf[buf][(wm * 32 + mt * 16 + l15) * LSTR + lh * 16]);
#pragma unroll
    for (int nt = 0; nt < 4; ++nt)
      bfr[nt] = ldfrag_b(&Bbuf[buf][(wn * 64 + nt * 16 + l15) * LSTR + lh * 16]);
#pragma unroll
    for (int mt = 0; mt < 2; ++mt)
#pragma unroll
      for (int nt = 0; nt < 4; ++nt)
        acc[mt][nt] = __builtin_amdgcn_wmma_f32_16x16x32_bf16(
            false, af[mt], false, bfr[nt], (short)0, acc[mt][nt], false, false);
    buf ^= 1;
  }

#pragma unroll
  for (int mt = 0; mt < 2; ++mt) {
    const int rbase = tm * 128 + wm * 32 + mt * 16 + lh * 8;  // D: row = r + 8*(l>>4)
#pragma unroll
    for (int nt = 0; nt < 4; ++nt) {
      const int col = tn * 128 + wn * 64 + nt * 16 + l15;
      const float bv = bias[col];
#pragma unroll
      for (int r = 0; r < 8; ++r)
        C[(size_t)(rbase + r) * N + col] = (OutT)(acc[mt][nt][r] + bv);
    }
  }
}

// ---- flash attention: block = 4 waves, one (b,h); wave = one 16-row q-tile --
#define KSTR 136   // K-tile padded row stride: 128 bf16 + 8 pad (272B)
#define VSTR 40    // V-tile padded row stride: 32 bf16 + 8 pad (80B)
__global__ __launch_bounds__(128)
void mla_flash_attn(const __bf16* __restrict__ Q,    // [B*S][DMODEL]
                    const __bf16* __restrict__ Kb,   // [B*S][DMODEL]
                    const __bf16* __restrict__ Vt,   // [B][H][DK][S]
                    const unsigned char* __restrict__ mask,  // [B][S]
                    __bf16* __restrict__ Ctx) {      // [B*S][DMODEL]
  __shared__ __align__(16) __bf16 Kbuf[2][32 * KSTR];   // 32 keys x 128 d
  __shared__ __align__(16) __bf16 Vbuf[2][128 * VSTR];  // 128 d x 32 keys
  __shared__ __align__(16) __bf16 Plds[4][16][32];
  const int lane = threadIdx.x & 31, wave = threadIdx.x >> 5;
  const int l15 = lane & 15, lh = lane >> 4;
  const int qt = blockIdx.x * 4 + wave;              // 0..127
  const int h = blockIdx.y, b = blockIdx.z;
  const float scale = 0.08838834764831845f;          // 1/sqrt(128)

  const __bf16* Kblk = Kb + (size_t)(b * SEQ) * DMODEL + h * DK;        // rows=keys
  const __bf16* Vblk = Vt + (size_t)((b * NHEADS + h) * DK) * SEQ;      // rows=d

  // Q A-fragments for all 4 K-steps, register-resident across the key loop
  const __bf16* qrow = Q + (size_t)(b * SEQ + qt * 16 + l15) * DMODEL + h * DK;
  v16bf qa[4];
#pragma unroll
  for (int s = 0; s < 4; ++s) qa[s] = ldfrag_a(qrow + s * 32 + lh * 16);

  v8f acc[8];
  v8f vz = {};
#pragma unroll
  for (int i = 0; i < 8; ++i) acc[i] = vz;
  float rmax[8], rsum[8];
#pragma unroll
  for (int r = 0; r < 8; ++r) { rmax[r] = -3.0e38f; rsum[r] = 0.0f; }

  if (wave == 0) {
    tdm_load_2d(lds_off_u32(&Kbuf[0][0]), Kblk, DK, 32, DMODEL, 5, 3); // 64DW rows
    tdm_load_2d(lds_off_u32(&Vbuf[0][0]), Vblk, 32, DK, SEQ, 3, 3);    // 16DW rows
  }
  int buf = 0;
  for (int kt = 0; kt < SEQ; kt += 32) {
    if (wave == 0) __builtin_amdgcn_s_wait_tensorcnt(0);
    __syncthreads();
    if (wave == 0 && kt + 32 < SEQ) {
      tdm_load_2d(lds_off_u32(&Kbuf[buf ^ 1][0]), Kblk + (size_t)(kt + 32) * DMODEL,
                  DK, 32, DMODEL, 5, 3);
      tdm_load_2d(lds_off_u32(&Vbuf[buf ^ 1][0]), Vblk + (kt + 32), 32, DK, SEQ, 3, 3);
    }

    // ---- scores: two 16-key sub-tiles x 4 K-steps of WMMA, K-tile from LDS ----
    v8f sc[2];
#pragma unroll
    for (int t = 0; t < 2; ++t) {
      v8f s = {};
#pragma unroll
      for (int ks = 0; ks < 4; ++ks) {
        v16bf kf = ldfrag_b(&Kbuf[buf][(t * 16 + l15) * KSTR + ks * 32 + lh * 16]);
        s = __builtin_amdgcn_wmma_f32_16x16x32_bf16(false, qa[ks], false, kf, (short)0, s, false, false);
      }
      const int key = kt + t * 16 + l15;
      const float mb = mask[b * SEQ + key] ? 0.0f : -3.0e38f;
#pragma unroll
      for (int r = 0; r < 8; ++r) s[r] = s[r] * scale + mb;
      sc[t] = s;
    }

    // ---- online softmax (row r+8*lh lives in one 16-lane half) ----
#pragma unroll
    for (int r = 0; r < 8; ++r) {
      float v = fmaxf(sc[0][r], sc[1][r]);
      v = fmaxf(v, __shfl_xor(v, 1));
      v = fmaxf(v, __shfl_xor(v, 2));
      v = fmaxf(v, __shfl_xor(v, 4));
      v = fmaxf(v, __shfl_xor(v, 8));
      const float nm = fmaxf(rmax[r], v);
      const float corr = __expf(rmax[r] - nm);
      rmax[r] = nm;
      const float e0 = __expf(sc[0][r] - nm);
      const float e1 = __expf(sc[1][r] - nm);
      float ps = e0 + e1;
      ps += __shfl_xor(ps, 1);
      ps += __shfl_xor(ps, 2);
      ps += __shfl_xor(ps, 4);
      ps += __shfl_xor(ps, 8);
      rsum[r] = rsum[r] * corr + ps;
#pragma unroll
      for (int nt = 0; nt < 8; ++nt) acc[nt][r] *= corr;
      Plds[wave][r + 8 * lh][l15]      = (__bf16)e0;   // D-layout -> row-major P
      Plds[wave][r + 8 * lh][16 + l15] = (__bf16)e1;
    }
    __syncthreads();

    // ---- PV: P(16x32) @ V(32x16) per 16-wide d tile, V-tile from LDS ----
    v16bf pf = ldfrag_a(&Plds[wave][l15][lh * 16]);
#pragma unroll
    for (int nt = 0; nt < 8; ++nt) {
      v16bf vf = ldfrag_b(&Vbuf[buf][(nt * 16 + l15) * VSTR + lh * 16]);
      acc[nt] = __builtin_amdgcn_wmma_f32_16x16x32_bf16(false, pf, false, vf, (short)0, acc[nt], false, false);
    }
    buf ^= 1;
  }

  const size_t qbase = (size_t)(b * SEQ + qt * 16 + lh * 8);
#pragma unroll
  for (int r = 0; r < 8; ++r) {
    const float inv = 1.0f / rsum[r];
#pragma unroll
    for (int nt = 0; nt < 8; ++nt)
      Ctx[(qbase + r) * DMODEL + h * DK + nt * 16 + l15] = (__bf16)(acc[nt][r] * inv);
  }
}

// ---- host orchestration -----------------------------------------------------
extern "C" void kernel_launch(void* const* d_in, const int* in_sizes, int n_in,
                              void* d_out, int out_size, void* d_ws, size_t ws_size,
                              hipStream_t stream) {
  const float* x       = (const float*)d_in[0];
  const unsigned char* mask = (const unsigned char*)d_in[1];
  const float* wq_k    = (const float*)d_in[2];
  const float* wq_b    = (const float*)d_in[3];
  const float* wkvd_k  = (const float*)d_in[4];
  const float* wkvd_b  = (const float*)d_in[5];
  const float* wku_k   = (const float*)d_in[6];
  const float* wku_b   = (const float*)d_in[7];
  const float* wvu_k   = (const float*)d_in[8];
  const float* wvu_b   = (const float*)d_in[9];
  const float* wo_k    = (const float*)d_in[10];
  const float* wo_b    = (const float*)d_in[11];
  float* out = (float*)d_out;

  char* ws = (char*)d_ws;
  size_t off = 0;
  auto alloc = [&](size_t bytes) -> char* {
    char* p = ws + off;
    off += (bytes + 255) & ~(size_t)255;
    return p;
  };
  __bf16* xbf   = (__bf16*)alloc((size_t)MROWS * DMODEL * 2);
  __bf16* wqT   = (__bf16*)alloc((size_t)DMODEL * DMODEL * 2);
  __bf16* wkvdT = (__bf16*)alloc((size_t)DLAT * DMODEL * 2);
  __bf16* wkuT  = (__bf16*)alloc((size_t)DMODEL * DLAT * 2);
  __bf16* wvuT  = (__bf16*)alloc((size_t)DMODEL * DLAT * 2);
  __bf16* woT   = (__bf16*)alloc((size_t)DMODEL * DMODEL * 2);
  __bf16* qbf   = (__bf16*)alloc((size_t)MROWS * DMODEL * 2);
  __bf16* kvl   = (__bf16*)alloc((size_t)MROWS * DLAT * 2);
  __bf16* kbf   = (__bf16*)alloc((size_t)MROWS * DMODEL * 2);
  __bf16* vbf   = (__bf16*)alloc((size_t)MROWS * DMODEL * 2);
  __bf16* vtb   = (__bf16*)alloc((size_t)MROWS * DMODEL * 2);
  __bf16* ctx   = (__bf16*)alloc((size_t)MROWS * DMODEL * 2);

  const int T = 256;
  { int n = MROWS * DMODEL; cvt_bf16_kernel<<<n / T, T, 0, stream>>>(x, xbf, n); }
  cvt_t_bf16_kernel<<<dim3(DMODEL / 32, DMODEL / 32), T, 0, stream>>>(wq_k,   wqT,   DMODEL, DMODEL);
  cvt_t_bf16_kernel<<<dim3(DLAT / 32,   DMODEL / 32), T, 0, stream>>>(wkvd_k, wkvdT, DMODEL, DLAT);
  cvt_t_bf16_kernel<<<dim3(DMODEL / 32, DLAT / 32),   T, 0, stream>>>(wku_k,  wkuT,  DLAT,   DMODEL);
  cvt_t_bf16_kernel<<<dim3(DMODEL / 32, DLAT / 32),   T, 0, stream>>>(wvu_k,  wvuT,  DLAT,   DMODEL);
  cvt_t_bf16_kernel<<<dim3(DMODEL / 32, DMODEL / 32), T, 0, stream>>>(wo_k,   woT,   DMODEL, DMODEL);

  gemm_bf16_wmma<__bf16><<<dim3(DMODEL / 128, MROWS / 128), T, 0, stream>>>(
      xbf, wqT, wq_b, qbf, MROWS, DMODEL, DMODEL);
  gemm_bf16_wmma<__bf16><<<dim3(DLAT / 128, MROWS / 128), T, 0, stream>>>(
      xbf, wkvdT, wkvd_b, kvl, MROWS, DLAT, DMODEL);
  gemm_bf16_wmma<__bf16><<<dim3(DMODEL / 128, MROWS / 128), T, 0, stream>>>(
      kvl, wkuT, wku_b, kbf, MROWS, DMODEL, DLAT);
  gemm_bf16_wmma<__bf16><<<dim3(DMODEL / 128, MROWS / 128), T, 0, stream>>>(
      kvl, wvuT, wvu_b, vbf, MROWS, DMODEL, DLAT);

  transpose_v_kernel<<<dim3(SEQ / 32, NHEADS * 4, B_SZ), T, 0, stream>>>(vbf, vtb);

  mla_flash_attn<<<dim3(SEQ / 64, NHEADS, B_SZ), 128, 0, stream>>>(qbf, kbf, vtb, mask, ctx);

  gemm_bf16_wmma<float><<<dim3(DMODEL / 128, MROWS / 128), T, 0, stream>>>(
      ctx, woT, wo_b, out, MROWS, DMODEL, DMODEL);
}
